// TemporalGCN_40097814675714
// MI455X (gfx1250) — matrix operand order repaired and checked by reading, not compiled, not run on hardware
//
#include <hip/hip_runtime.h>
#include <hip/hip_bf16.h>

#define BATCH 32
#define TLEN  512
#define CN    22      // real node count
#define CP    32      // padded node count
#define NF    192     // FREQ
#define HID   128

typedef __attribute__((ext_vector_type(16))) __bf16 bf16x16;
typedef __attribute__((ext_vector_type(8)))  __bf16 bf16x8;
typedef __attribute__((ext_vector_type(8)))  float  f32x8;
typedef __attribute__((ext_vector_type(4)))  float  f32x4;

__device__ __forceinline__ __bf16 to_bf(float f) { return (__bf16)f; }

// ---- A-side fragment (16x32 bf16) from row-major LDS matrix, stride S elems.
// ISA layout: lane holds row m=lane&15; lanes<16: K = kb..kb+7 then kb+16..kb+23
// with kb = 8*(lane>=16).
__device__ __forceinline__ bf16x16 lds_load_a(const __bf16* p, int S, int row0,
                                              int k0, int lane) {
  int m  = row0 + (lane & 15);
  int kb = k0 + ((lane >> 4) << 3);
  const __bf16* r = p + m * S + kb;
  bf16x8 lo = *(const bf16x8*)(r);
  bf16x8 hi = *(const bf16x8*)(r + 16);
  bf16x16 a;
#pragma unroll
  for (int j = 0; j < 8; ++j) { a[j] = lo[j]; a[8 + j] = hi[j]; }
  return a;
}

// ---- A-side fragment loaded from GLOBAL f32 row-major (weights), cvt->bf16.
__device__ __forceinline__ bf16x16 gld_a_f32(const float* __restrict__ p, int S,
                                             int row0, int k0, int lane) {
  int m  = row0 + (lane & 15);
  int kb = k0 + ((lane >> 4) << 3);
  const float* r = p + (size_t)m * S + kb;
  f32x4 f0 = *(const f32x4*)(r);
  f32x4 f1 = *(const f32x4*)(r + 4);
  f32x4 f2 = *(const f32x4*)(r + 16);
  f32x4 f3 = *(const f32x4*)(r + 20);
  bf16x16 a;
#pragma unroll
  for (int j = 0; j < 4; ++j) {
    a[j]      = to_bf(f0[j]);
    a[4 + j]  = to_bf(f1[j]);
    a[8 + j]  = to_bf(f2[j]);
    a[12 + j] = to_bf(f3[j]);
  }
  return a;
}

// ---- B-side fragment (32x16 bf16) from Bt (= B transposed) row-major LDS.
// ISA layout: lane holds column n=lane&15; K = kb..kb+7 then kb+16..kb+23, kb = k0+8*(lane>=16).
__device__ __forceinline__ bf16x16 lds_load_b(const __bf16* p, int S, int k0,
                                              int n0, int lane) {
  int n  = n0 + (lane & 15);
  int kb = k0 + ((lane >> 4) << 3);
  const __bf16* r = p + n * S + kb;
  bf16x8 lo = *(const bf16x8*)(r);
  bf16x8 hi = *(const bf16x8*)(r + 16);
  bf16x16 b;
#pragma unroll
  for (int j = 0; j < 8; ++j) { b[j] = lo[j]; b[8 + j] = hi[j]; }
  return b;
}

// ---- Store D transposed (dest[n][m], row-major, stride S) as bf16.
// Lane writes row n, 8 contiguous columns m0+8*(lane>=16)+0..7 -> one b128 store.
__device__ __forceinline__ void st_dT_bf16(__bf16* p, int S, int n0, int m0,
                                           f32x8 d, int lane) {
  __bf16* r = p + (n0 + (lane & 15)) * S + m0 + ((lane >> 4) << 3);
  bf16x8 v;
#pragma unroll
  for (int j = 0; j < 8; ++j) v[j] = to_bf(d[j]);
  *(bf16x8*)r = v;
}

// ---- Store D non-transposed (dest[m][n], row-major, stride S) as bf16.
__device__ __forceinline__ void st_d_bf16(__bf16* p, int S, int m0, int n0,
                                          f32x8 d, int lane) {
  int n  = n0 + (lane & 15);
  int mb = m0 + ((lane >> 4) << 3);
#pragma unroll
  for (int v = 0; v < 8; ++v) p[(mb + v) * S + n] = to_bf(d[v]);
}

// ---- Store D transposed (dest[n][m]) as f32.
__device__ __forceinline__ void st_dT_f32(float* p, int S, int n0, int m0,
                                          f32x8 d, int lane) {
  float* r = p + (n0 + (lane & 15)) * S + m0 + ((lane >> 4) << 3);
  f32x4 lo, hi;
#pragma unroll
  for (int j = 0; j < 4; ++j) { lo[j] = d[j]; hi[j] = d[4 + j]; }
  *(f32x4*)(r)     = lo;
  *(f32x4*)(r + 4) = hi;
}

#define WMMA_BF16(a, b, c)                                                     \
  __builtin_amdgcn_wmma_f32_16x16x32_bf16(false, (a), false, (b), (short)0,    \
                                          (c), false, false)

__global__ void __launch_bounds__(256)
tgcn_wmma_kernel(const float* __restrict__ x, const float* __restrict__ A,
                 const float* __restrict__ W0, const float* __restrict__ b0,
                 const float* __restrict__ W1, const float* __restrict__ b1,
                 float* __restrict__ out) {
  __shared__ __bf16 sAdj[CP * CP];    //  2 KB  adjacency, padded, row-major (= Bt of A^T)
  __shared__ __bf16 sXT[NF * CP];     // 12 KB  x transposed [f][c]
  __shared__ __bf16 sH1[CP * NF];     // 12 KB  h1 [c][f]   (Bt for stage 2)
  __shared__ __bf16 sH2T[HID * CP];   //  8 KB  h2^T [h][c] (A-side for stage 3)
  __shared__ __bf16 sH3[CP * HID];    //  8 KB  h3 [c][h]   (Bt for stage 4)
  __shared__ float  sH4[CP * HID];    // 16 KB  h4 [c][h]   f32 for final mean

  const int bt   = blockIdx.x;        // 0 .. B*T-1
  const int tid  = threadIdx.x;
  const int lane = tid & 31;
  const int w    = tid >> 5;          // wave 0..7

  const float* xg = x + (size_t)bt * (CN * NF);
  const float* Ag = A + (size_t)bt * (CN * CN);

  // ---- Stage 0: stage adjacency (zero-padded 32x32) and x^T into LDS as bf16.
  for (int i = tid; i < CP * CP; i += 256) {
    int r = i >> 5, c = i & 31;
    float v = (r < CN && c < CN) ? Ag[r * CN + c] : 0.0f;
    sAdj[i] = to_bf(v);
  }
  for (int i = tid; i < CN * NF; i += 256) {
    int c = i / NF, f = i % NF;
    sXT[f * CP + c] = to_bf(xg[i]);
  }
  for (int i = tid; i < NF * (CP - CN); i += 256) {
    int f = i / (CP - CN), c = CN + i % (CP - CN);
    sXT[f * CP + c] = to_bf(0.0f);
  }
  __syncthreads();

  // ---- Stage 1: h1^T = x^T @ A^T   (M=192, K=32, N=32). Store D transposed.
  for (int tile = w; tile < NF / 16; tile += 8) {
    bf16x16 a = lds_load_a(sXT, CP, tile * 16, 0, lane);
#pragma unroll
    for (int nt = 0; nt < 2; ++nt) {
      bf16x16 bm = lds_load_b(sAdj, CP, 0, nt * 16, lane);
      f32x8 d = {};
      d = WMMA_BF16(a, bm, d);
      st_dT_bf16(sH1, NF, nt * 16, tile * 16, d, lane);
    }
  }
  __syncthreads();

  // ---- Stage 2: h2^T = relu(W0 @ h1^T + b0)  (M=128, K=192, N=32).
  {
    f32x8 d0 = {}, d1 = {};
    for (int k0 = 0; k0 < NF; k0 += 32) {
      bf16x16 a   = gld_a_f32(W0, NF, w * 16, k0, lane);
      bf16x16 bm0 = lds_load_b(sH1, NF, k0, 0, lane);
      bf16x16 bm1 = lds_load_b(sH1, NF, k0, 16, lane);
      d0 = WMMA_BF16(a, bm0, d0);
      d1 = WMMA_BF16(a, bm1, d1);
    }
    const float* bp = b0 + w * 16 + ((lane >> 4) << 3);
    f32x4 blo = *(const f32x4*)bp;
    f32x4 bhi = *(const f32x4*)(bp + 4);
#pragma unroll
    for (int v = 0; v < 4; ++v) {
      d0[v]     = fmaxf(d0[v] + blo[v], 0.0f);
      d0[4 + v] = fmaxf(d0[4 + v] + bhi[v], 0.0f);
      d1[v]     = fmaxf(d1[v] + blo[v], 0.0f);
      d1[4 + v] = fmaxf(d1[4 + v] + bhi[v], 0.0f);
    }
    st_d_bf16(sH2T, CP, w * 16, 0, d0, lane);
    st_d_bf16(sH2T, CP, w * 16, 16, d1, lane);
  }
  __syncthreads();

  // ---- Stage 3: h3^T = h2^T @ A^T  (M=128, K=32, N=32). Store D transposed.
  {
    bf16x16 a = lds_load_a(sH2T, CP, w * 16, 0, lane);
#pragma unroll
    for (int nt = 0; nt < 2; ++nt) {
      bf16x16 bm = lds_load_b(sAdj, CP, 0, nt * 16, lane);
      f32x8 d = {};
      d = WMMA_BF16(a, bm, d);
      st_dT_bf16(sH3, HID, nt * 16, w * 16, d, lane);
    }
  }
  __syncthreads();

  // ---- Stage 4: h4^T = relu(W1 @ h3^T + b1)  (M=128, K=128, N=32).
  {
    f32x8 d0 = {}, d1 = {};
    for (int k0 = 0; k0 < HID; k0 += 32) {
      bf16x16 a   = gld_a_f32(W1, HID, w * 16, k0, lane);
      bf16x16 bm0 = lds_load_b(sH3, HID, k0, 0, lane);
      bf16x16 bm1 = lds_load_b(sH3, HID, k0, 16, lane);
      d0 = WMMA_BF16(a, bm0, d0);
      d1 = WMMA_BF16(a, bm1, d1);
    }
    const float* bp = b1 + w * 16 + ((lane >> 4) << 3);
    f32x4 blo = *(const f32x4*)bp;
    f32x4 bhi = *(const f32x4*)(bp + 4);
#pragma unroll
    for (int v = 0; v < 4; ++v) {
      d0[v]     = fmaxf(d0[v] + blo[v], 0.0f);
      d0[4 + v] = fmaxf(d0[4 + v] + bhi[v], 0.0f);
      d1[v]     = fmaxf(d1[v] + blo[v], 0.0f);
      d1[4 + v] = fmaxf(d1[4 + v] + bhi[v], 0.0f);
    }
    st_dT_f32(sH4, HID, 0, w * 16, d0, lane);
    st_dT_f32(sH4, HID, 16, w * 16, d1, lane);
  }
  __syncthreads();

  // ---- Stage 5: mean over the 22 real nodes -> out[bt][h].
  if (tid < HID) {
    float acc = 0.0f;
#pragma unroll 11
    for (int c = 0; c < CN; ++c) acc += sH4[c * HID + tid];
    out[(size_t)bt * HID + tid] = acc * (1.0f / (float)CN);
  }
}

extern "C" void kernel_launch(void* const* d_in, const int* in_sizes, int n_in,
                              void* d_out, int out_size, void* d_ws, size_t ws_size,
                              hipStream_t stream) {
  (void)in_sizes; (void)n_in; (void)out_size; (void)d_ws; (void)ws_size;
  const float* x  = (const float*)d_in[0];
  const float* A  = (const float*)d_in[1];
  const float* W0 = (const float*)d_in[2];
  const float* b0 = (const float*)d_in[3];
  const float* W1 = (const float*)d_in[4];
  const float* b1 = (const float*)d_in[5];
  float* out = (float*)d_out;

  dim3 grid(BATCH * TLEN);  // one workgroup per (b, t)
  dim3 block(256);          // 8 wave32s
  tgcn_wmma_kernel<<<grid, block, 0, stream>>>(x, A, W0, b0, W1, b1, out);
}
